// MLPDecoder_23227183136845
// MI455X (gfx1250) — matrix operand order repaired and checked by reading, not compiled, not run on hardware
//
#include <hip/hip_runtime.h>
#include <hip/hip_bf16.h>
#include <math.h>

// Problem constants (from reference)
#define DD   128
#define EE   500000
#define MM   5
#define TEMP_INV 10.0f   // 1 / 0.1

typedef __attribute__((ext_vector_type(16))) __bf16 v16bf;
typedef __attribute__((ext_vector_type(8)))  float  v8f;

union Frag16 {
    v16bf          v;
    unsigned short u[16];
    unsigned int   d[8];
    uint4          q[2];
};

// Cold path (weights preload): proper round-to-nearest-even.
static __device__ __forceinline__ unsigned short f2bf_rne(float f) {
    unsigned int b = __float_as_uint(f);
    b += 0x7FFFu + ((b >> 16) & 1u);
    return (unsigned short)(b >> 16);
}

// Hot path: pack two floats into packed bf16 with round-half-up:
// 2x v_add_u32 + 1x v_perm_b32 (selector picks the two high halves).
static __device__ __forceinline__ unsigned int pkbf(float lo, float hi) {
    unsigned int bl = __float_as_uint(lo) + 0x8000u;
    unsigned int bh = __float_as_uint(hi) + 0x8000u;
    return __builtin_amdgcn_perm(bh, bl, 0x07060302u);
}

static __device__ __forceinline__ unsigned short f2bf_fast(float f) {
    return (unsigned short)((__float_as_uint(f) + 0x8000u) >> 16);
}

// ---------------------------------------------------------------------------
// Kernel 1: per 16-edge tile:
//   concat-gather -> bf16 WMMA GEMM (16x256 @ 256x128) -> ReLU
//   -> bf16 WMMA GEMM (16x128 @ 128x64) -> per-row InfoNCE partial loss
// One wave per tile, 8 waves (256 threads) per block.
// ---------------------------------------------------------------------------
__global__ void __launch_bounds__(256)
mlp_infonce_kernel(const float* __restrict__ ufeat, const float* __restrict__ ifeat,
                   const float* __restrict__ review, const float* __restrict__ negk,
                   const float* __restrict__ W1, const float* __restrict__ W2,
                   const int* __restrict__ srcIdx, const int* __restrict__ dstIdx,
                   float* __restrict__ blockPartial, int numTiles)
{
    extern __shared__ char smem[];
    unsigned short* sW1     = (unsigned short*)smem;         // 32768 bf16 (64KB), B-frag order
    unsigned short* sW2     = sW1 + 32768;                   //  8192 bf16 (16KB), B-frag order
    unsigned short* sHidAll = sW2 + 8192;                    // 8 waves * 16x128 bf16 (32KB)
    float*          sOutAll = (float*)(sHidAll + 8 * 2048);  // 8 waves * 16x64 f32  (32KB)
    float*          sPartial = sOutAll + 8 * 1024;           // 8 floats

    const int tid  = threadIdx.x;
    const int lane = tid & 31;
    const int wave = tid >> 5;

    // ---- Preload W1/W2 into LDS, swizzled into WMMA B-fragment order ----
    // B layout (32x16 bf16): lane = (n&15) + 16*((kin>>3)&1), vector elem j = (kin&7) | ((kin>>4)<<3).
    for (int e = tid; e < 256 * 128; e += 256) {
        int k = e >> 7, n = e & 127;
        int nt = n >> 4, lo = n & 15;
        int kc = k >> 5, kin = k & 31;
        int half = (kin >> 3) & 1, j = (kin & 7) | ((kin >> 4) << 3);
        sW1[(((nt * 8 + kc) * 32) + (lo + (half << 4))) * 16 + j] = f2bf_rne(W1[e]);
    }
    for (int e = tid; e < 128 * 64; e += 256) {
        int k = e >> 6, n = e & 63;
        int nt = n >> 4, lo = n & 15;
        int kc = k >> 5, kin = k & 31;
        int half = (kin >> 3) & 1, j = (kin & 7) | ((kin >> 4) << 3);
        sW2[(((nt * 4 + kc) * 32) + (lo + (half << 4))) * 16 + j] = f2bf_rne(W2[e]);
    }
    __syncthreads();

    float partial = 0.0f;
    const int tile = blockIdx.x * 8 + wave;

    if (tile < numTiles) {
        unsigned short* sHid = sHidAll + wave * 2048;  // 16 x 128 bf16, row-major
        float*          sOut = sOutAll + wave * 1024;  // 16 x 64  f32,  row-major

        const int row   = lane & 15;                  // A-matrix row / D-tile column
        const int kb    = (lane < 16) ? 0 : 8;        // A-layout K base (ISA 7.12.2)
        const int moff  = (lane < 16) ? 0 : 8;        // D-layout M base
        const int edge0 = tile * 16;

        const float* pu = ufeat + (size_t)srcIdx[edge0 + row] * DD;
        const float* pi = ifeat + (size_t)dstIdx[edge0 + row] * DD;

        // ---- A fragments for layer 1: 8 chunks of K=32 over concat(u,i) ----
        Frag16 afrag[8];
#pragma unroll
        for (int kc = 0; kc < 8; ++kc) {
            const float* rp = (kc < 4) ? (pu + kc * 32) : (pi + (kc - 4) * 32);
            float4 a0 = *(const float4*)(rp + kb);
            float4 a1 = *(const float4*)(rp + kb + 4);
            float4 a2 = *(const float4*)(rp + 16 + kb);
            float4 a3 = *(const float4*)(rp + 16 + kb + 4);
            afrag[kc].d[0] = pkbf(a0.x, a0.y);
            afrag[kc].d[1] = pkbf(a0.z, a0.w);
            afrag[kc].d[2] = pkbf(a1.x, a1.y);
            afrag[kc].d[3] = pkbf(a1.z, a1.w);
            afrag[kc].d[4] = pkbf(a2.x, a2.y);
            afrag[kc].d[5] = pkbf(a2.z, a2.w);
            afrag[kc].d[6] = pkbf(a3.x, a3.y);
            afrag[kc].d[7] = pkbf(a3.z, a3.w);
        }

        // ---- Layer 1: hidden[16x128] = relu(A[16x256] @ W1[256x128]) ----
#pragma unroll
        for (int nt = 0; nt < 8; ++nt) {
            v8f acc = {0.f, 0.f, 0.f, 0.f, 0.f, 0.f, 0.f, 0.f};
#pragma unroll
            for (int kc = 0; kc < 8; ++kc) {
                Frag16 bfr;
                const uint4* bp = (const uint4*)(sW1 + (((nt * 8 + kc) * 32) + lane) * 16);
                bfr.q[0] = bp[0];
                bfr.q[1] = bp[1];
                acc = __builtin_amdgcn_wmma_f32_16x16x32_bf16(
                    false, afrag[kc].v, false, bfr.v, (short)0, acc, false, false);
            }
            const int col = nt * 16 + row;
#pragma unroll
            for (int v = 0; v < 8; ++v) {
                float x = acc[v];
                x = x > 0.f ? x : 0.f;                       // ReLU
                sHid[(moff + v) * 128 + col] = f2bf_fast(x);
            }
        }
        asm volatile("s_wait_dscnt 0" ::: "memory");         // wave-local LDS RAW

        // ---- Layer-2 A fragments (16x32 per chunk) from LDS hidden ----
        Frag16 a2frag[4];
#pragma unroll
        for (int kc = 0; kc < 4; ++kc) {
            a2frag[kc].q[0] = *(const uint4*)(sHid + row * 128 + kc * 32 + kb);
            a2frag[kc].q[1] = *(const uint4*)(sHid + row * 128 + kc * 32 + 16 + kb);
        }

        // ---- Layer 2: out[16x64] = hidden[16x128] @ W2[128x64] ----
#pragma unroll
        for (int nt = 0; nt < 4; ++nt) {
            v8f acc = {0.f, 0.f, 0.f, 0.f, 0.f, 0.f, 0.f, 0.f};
#pragma unroll
            for (int kc = 0; kc < 4; ++kc) {
                Frag16 bfr;
                const uint4* bp = (const uint4*)(sW2 + (((nt * 4 + kc) * 32) + lane) * 16);
                bfr.q[0] = bp[0];
                bfr.q[1] = bp[1];
                acc = __builtin_amdgcn_wmma_f32_16x16x32_bf16(
                    false, a2frag[kc].v, false, bfr.v, (short)0, acc, false, false);
            }
            const int col = nt * 16 + row;
#pragma unroll
            for (int v = 0; v < 8; ++v)
                sOut[(moff + v) * 64 + col] = acc[v];
        }
        asm volatile("s_wait_dscnt 0" ::: "memory");

        // ---- InfoNCE row phase: 2 lanes per row, 32 dims each ----
        const int r  = lane >> 1;
        const int hh = lane & 1;
        const size_t edge = (size_t)edge0 + r;

        float qv[32];
        const float4* q4 = (const float4*)(sOut + r * 64 + hh * 32);
        float qsq = 0.f;
#pragma unroll
        for (int i = 0; i < 8; ++i) {
            float4 t = q4[i];
            qv[4*i] = t.x; qv[4*i+1] = t.y; qv[4*i+2] = t.z; qv[4*i+3] = t.w;
            qsq += t.x*t.x + t.y*t.y + t.z*t.z + t.w*t.w;
        }
        qsq += __shfl_xor(qsq, 1);

        const float4* p4 = (const float4*)(review + edge * 64 + hh * 32);
        float psq = 0.f, qd = 0.f;
#pragma unroll
        for (int i = 0; i < 8; ++i) {
            float4 t = p4[i];
            psq += t.x*t.x + t.y*t.y + t.z*t.z + t.w*t.w;
            qd  += qv[4*i]*t.x + qv[4*i+1]*t.y + qv[4*i+2]*t.z + qv[4*i+3]*t.w;
        }
        psq += __shfl_xor(psq, 1);
        qd  += __shfl_xor(qd, 1);

        const float invq = rsqrtf(qsq);
        const float l0   = qd * invq * rsqrtf(psq) * TEMP_INV;
        float mx = l0;
        float ln[MM];
#pragma unroll
        for (int m = 0; m < MM; ++m) {
            const float4* n4 = (const float4*)(negk + (edge * MM + m) * 64 + hh * 32);
            float nsq = 0.f, nd = 0.f;
#pragma unroll
            for (int i = 0; i < 8; ++i) {
                float4 t = n4[i];
                nsq += t.x*t.x + t.y*t.y + t.z*t.z + t.w*t.w;
                nd  += qv[4*i]*t.x + qv[4*i+1]*t.y + qv[4*i+2]*t.z + qv[4*i+3]*t.w;
            }
            nsq += __shfl_xor(nsq, 1);
            nd  += __shfl_xor(nd, 1);
            float lm = nd * invq * rsqrtf(nsq) * TEMP_INV;
            ln[m] = lm;
            mx = fmaxf(mx, lm);
        }
        float ssum = __expf(l0 - mx);
#pragma unroll
        for (int m = 0; m < MM; ++m) ssum += __expf(ln[m] - mx);
        if (hh == 0) partial = mx + __logf(ssum) - l0;   // logsumexp - logits[0]
    }

    // ---- Deterministic in-block reduction ----
#pragma unroll
    for (int off = 16; off > 0; off >>= 1) partial += __shfl_xor(partial, off);
    if (lane == 0) sPartial[wave] = partial;
    __syncthreads();
    if (tid == 0) {
        float s = 0.f;
#pragma unroll
        for (int w = 0; w < 8; ++w) s += sPartial[w];
        blockPartial[blockIdx.x] = s;
    }
}

// ---------------------------------------------------------------------------
// Kernel 2: deterministic final reduction -> loss = sum / E
// ---------------------------------------------------------------------------
__global__ void __launch_bounds__(256)
infonce_reduce_kernel(const float* __restrict__ ws, int n, float* __restrict__ out)
{
    __shared__ float s[256];
    float acc = 0.f;
    for (int i = threadIdx.x; i < n; i += 256) acc += ws[i];
    s[threadIdx.x] = acc;
    __syncthreads();
    for (int off = 128; off > 0; off >>= 1) {
        if (threadIdx.x < off) s[threadIdx.x] += s[threadIdx.x + off];
        __syncthreads();
    }
    if (threadIdx.x == 0) out[0] = s[0] * (1.0f / (float)EE);
}

extern "C" void kernel_launch(void* const* d_in, const int* in_sizes, int n_in,
                              void* d_out, int out_size, void* d_ws, size_t ws_size,
                              hipStream_t stream) {
    (void)in_sizes; (void)n_in; (void)out_size; (void)ws_size;
    const float* ufeat  = (const float*)d_in[0];
    const float* ifeat  = (const float*)d_in[1];
    const float* review = (const float*)d_in[2];
    const float* negk   = (const float*)d_in[3];
    const float* W1     = (const float*)d_in[4];
    const float* W2     = (const float*)d_in[5];
    const int*   srcI   = (const int*)d_in[6];
    const int*   dstI   = (const int*)d_in[7];
    float* out = (float*)d_out;
    float* ws  = (float*)d_ws;

    const int numTiles = EE / 16;              // 31250 tiles of 16 edges
    const int blocks   = (numTiles + 7) / 8;   // 8 waves (tiles) per block
    // LDS: W1(64KB) + W2(16KB) + hidden(32KB) + out(32KB) + partials
    const size_t shmem = (size_t)(32768 + 8192 + 8 * 2048) * sizeof(unsigned short)
                       + (size_t)(8 * 1024 + 8) * sizeof(float);

    mlp_infonce_kernel<<<blocks, 256, shmem, stream>>>(
        ufeat, ifeat, review, negk, W1, W2, srcI, dstI, ws, numTiles);
    infonce_reduce_kernel<<<1, 256, 0, stream>>>(ws, blocks, out);
}
